// BasicRNN_3100966387811
// MI455X (gfx1250) — matrix operand order repaired and checked by reading, not compiled, not run on hardware
//
#include <hip/hip_runtime.h>

typedef __attribute__((ext_vector_type(16))) __bf16 v16bf;
typedef __attribute__((ext_vector_type(8)))  float  v8f;

#define TB      16      // batch rows per workgroup
#define SEQ     64
#define HIDN    32
#define GATES   128     // 4*H
#define NXI     4
#define NSFC_   5
#define THREADS 256     // 8 waves of 32

union BF16Frag { v16bf v; uint4 u[2]; };

// ---- native CDNA5 transcendental tanh (v_tanh_f32); host-safe fallback ----
__device__ __forceinline__ float fast_tanhf(float x) {
#if defined(__HIP_DEVICE_COMPILE__) && __has_builtin(__builtin_amdgcn_tanhf)
    return __builtin_amdgcn_tanhf(x);
#else
    return tanhf(x);
#endif
}
__device__ __forceinline__ float fast_sigmoidf(float x) {
    return 0.5f + 0.5f * fast_tanhf(0.5f * x);
}

__global__ __launch_bounds__(THREADS, 1)
void lstm2_fused_kernel(
    const float* __restrict__ gx,                              // [B,S,4]
    const float* __restrict__ gsfc,                            // [B,5]
    const float* __restrict__ w_sfc1, const float* __restrict__ b_sfc1,
    const float* __restrict__ w_sfc2, const float* __restrict__ b_sfc2,
    const float* __restrict__ w_ih1,  const float* __restrict__ w_hh1,
    const float* __restrict__ b_ih1,  const float* __restrict__ b_hh1,
    const float* __restrict__ w_ih2,  const float* __restrict__ w_hh2,
    const float* __restrict__ b_ih2,  const float* __restrict__ b_hh2,
    const float* __restrict__ w_out,  const float* __restrict__ b_out,
    float* __restrict__ gy)                                    // [B,S,1]
{
    // ---- LDS (tile-resident state; ~120 KB) ----
    __shared__ __attribute__((aligned(16))) __bf16 sOut[TB][SEQ*HIDN + 24]; // rnn1 output
    __shared__ float  sX[TB][SEQ*NXI];
    __shared__ float  sGate[TB][GATES];
    __shared__ __attribute__((aligned(16))) __bf16 sH[TB][40];  // padded row
    __shared__ __attribute__((aligned(16))) __bf16 sWhh1[GATES*HIDN];
    __shared__ __attribute__((aligned(16))) __bf16 sWhh2[GATES*HIDN];
    __shared__ __attribute__((aligned(16))) __bf16 sWih2[GATES*HIDN];
    __shared__ float  sWih1[GATES][NXI];
    __shared__ float  sB1[GATES], sB2[GATES];
    __shared__ float  sWsfc1[HIDN][8], sWsfc2[HIDN][8];
    __shared__ float  sBsfc1[HIDN], sBsfc2[HIDN];
    __shared__ float  sWout[HIDN];
    __shared__ float  sSfc[TB][8];

    const int tid   = threadIdx.x;
    const int lane  = tid & 31;
    const int wave  = tid >> 5;            // gate-column tile n0 = wave*16
    const int bbase = blockIdx.x * TB;

    // ---- stage inputs & weights into LDS ----
    for (int i = tid; i < TB*SEQ*NXI; i += THREADS)
        ((float*)sX)[i] = gx[(size_t)bbase*SEQ*NXI + i];
    for (int i = tid; i < GATES*HIDN; i += THREADS) {
        sWhh1[i] = (__bf16)w_hh1[i];
        sWhh2[i] = (__bf16)w_hh2[i];
        sWih2[i] = (__bf16)w_ih2[i];
    }
    for (int i = tid; i < GATES*NXI; i += THREADS)
        ((float*)sWih1)[i] = w_ih1[i];
    for (int i = tid; i < GATES; i += THREADS) {
        sB1[i] = b_ih1[i] + b_hh1[i];
        sB2[i] = b_ih2[i] + b_hh2[i];
    }
    for (int i = tid; i < HIDN*NSFC_; i += THREADS) {
        int r = i / NSFC_, c = i % NSFC_;
        sWsfc1[r][c] = w_sfc1[i];
        sWsfc2[r][c] = w_sfc2[i];
    }
    for (int i = tid; i < HIDN; i += THREADS) {
        sBsfc1[i] = b_sfc1[i];
        sBsfc2[i] = b_sfc2[i];
        sWout[i]  = w_out[i];
    }
    for (int i = tid; i < TB*NSFC_; i += THREADS) {
        int b = i / NSFC_, k = i % NSFC_;
        sSfc[b][k] = gsfc[(size_t)(bbase + b)*NSFC_ + k];
    }
    __syncthreads();

    // ---- hoisted B-fragments (weights loop-invariant; kept in VGPRs) ----
    // B layout: lane N = gate column (lane&15), elems 0..7 -> K = kb..kb+7, elems 8..15 -> K = 16+kb..16+kb+7, kb = 8*(lane>>4)
    const int gcol = wave*16 + (lane & 15);
    const int kb2  = (lane >> 4) * 8;
    BF16Frag fWhh1, fWhh2, fWih2;
    {
        const uint4* p1 = (const uint4*)&sWhh1[gcol*HIDN + kb2];
        fWhh1.u[0] = p1[0]; fWhh1.u[1] = p1[2];
        const uint4* p2 = (const uint4*)&sWhh2[gcol*HIDN + kb2];
        fWhh2.u[0] = p2[0]; fWhh2.u[1] = p2[2];
        const uint4* p3 = (const uint4*)&sWih2[gcol*HIDN + kb2];
        fWih2.u[0] = p3[0]; fWih2.u[1] = p3[2];
    }

    // ---- elementwise-stage mapping: wave w owns batch rows {w, w+8}, lane = hid ----
    const int cb0  = tid >> 5;       // == wave
    const int cb1  = cb0 + 8;
    const int chid = tid & 31;
    float creg[2];

    // per-thread loop-invariant registers (barriers can't evict these)
    float wih1r[4][4], b1r[4], b2r[4];
    #pragma unroll
    for (int gt = 0; gt < 4; ++gt) {
        const int g = chid + 32*gt;
        b1r[gt] = sB1[g];
        b2r[gt] = sB2[g];
        #pragma unroll
        for (int k = 0; k < NXI; ++k) wih1r[gt][k] = sWih1[g][k];
    }
    const float woutc = sWout[chid];
    const float bout  = b_out[0];          // uniform scalar load

    // ---- surface MLP -> h0, c0 of rnn1 ----
    #pragma unroll
    for (int p = 0; p < 2; ++p) {
        const int b = p ? cb1 : cb0;
        float a1 = sBsfc1[chid], a2 = sBsfc2[chid];
        #pragma unroll
        for (int k = 0; k < NSFC_; ++k) {
            a1 += sSfc[b][k] * sWsfc1[chid][k];
            a2 += sSfc[b][k] * sWsfc2[chid][k];
        }
        sH[b][chid] = (__bf16)fast_tanhf(a1);
        creg[p]     = fast_tanhf(a2);
    }
    __syncthreads();

    // A-fragment addressing (16x32 bf16): lane M = lane&15,
    // elems 0..7 -> K = kb..kb+7 ; elems 8..15 -> K = 16+kb..16+kb+7, kb = 8*(lane>>4)
    const int am  = lane & 15;
    const int akb = (lane >> 4) * 8;
    const int cM  = 8 * (lane >> 4);           // C-tile row base for this lane half
    const int cN  = wave*16 + (lane & 15);     // C-tile column in sGate

    // ================= rnn1 (time-reversed input) =================
    for (int s = 0; s < SEQ; ++s) {
        BF16Frag fA;
        fA.u[0] = *(const uint4*)&sH[am][akb];
        fA.u[1] = *(const uint4*)&sH[am][16 + akb];
        v8f acc = {0.f,0.f,0.f,0.f,0.f,0.f,0.f,0.f};
        acc = __builtin_amdgcn_wmma_f32_16x16x32_bf16(
                  false, fA.v, false, fWhh1.v, (short)0, acc, false, false);
        #pragma unroll
        for (int r = 0; r < 8; ++r)
            sGate[cM + r][cN] = acc[r];
        __syncthreads();

        const int tin = SEQ - 1 - s;           // flipped time index
        #pragma unroll
        for (int p = 0; p < 2; ++p) {
            const int b = p ? cb1 : cb0;
            const float4 xv = *(const float4*)&sX[b][tin*NXI];
            float gv[4];
            #pragma unroll
            for (int gt = 0; gt < 4; ++gt) {
                const int g = chid + 32*gt;
                gv[gt] = sGate[b][g] + b1r[gt]
                       + xv.x*wih1r[gt][0] + xv.y*wih1r[gt][1]
                       + xv.z*wih1r[gt][2] + xv.w*wih1r[gt][3];
            }
            const float iv = fast_sigmoidf(gv[0]);
            const float fv = fast_sigmoidf(gv[1]);
            const float gg = fast_tanhf(gv[2]);
            const float ov = fast_sigmoidf(gv[3]);
            const float c  = fv*creg[p] + iv*gg;
            creg[p] = c;
            const float h  = ov * fast_tanhf(c);
            const __bf16 hb = (__bf16)h;
            sH[b][chid] = hb;
            sOut[b][tin*HIDN + chid] = hb;     // flipped-back rnn1 output
        }
        __syncthreads();
    }

    // ================= rnn2 (zero initial state) =================
    creg[0] = 0.f; creg[1] = 0.f;
    sH[cb0][chid] = (__bf16)0.f;
    sH[cb1][chid] = (__bf16)0.f;
    __syncthreads();

    for (int t = 0; t < SEQ; ++t) {
        BF16Frag fAx, fAh;
        fAx.u[0] = *(const uint4*)&sOut[am][t*HIDN + akb];
        fAx.u[1] = *(const uint4*)&sOut[am][t*HIDN + 16 + akb];
        fAh.u[0] = *(const uint4*)&sH[am][akb];
        fAh.u[1] = *(const uint4*)&sH[am][16 + akb];
        v8f acc = {0.f,0.f,0.f,0.f,0.f,0.f,0.f,0.f};
        acc = __builtin_amdgcn_wmma_f32_16x16x32_bf16(     // input projection
                  false, fAx.v, false, fWih2.v, (short)0, acc, false, false);
        acc = __builtin_amdgcn_wmma_f32_16x16x32_bf16(     // recurrence, chained
                  false, fAh.v, false, fWhh2.v, (short)0, acc, false, false);
        #pragma unroll
        for (int r = 0; r < 8; ++r)
            sGate[cM + r][cN] = acc[r];
        __syncthreads();

        float hh[2];
        #pragma unroll
        for (int p = 0; p < 2; ++p) {
            const int b = p ? cb1 : cb0;
            float gv[4];
            #pragma unroll
            for (int gt = 0; gt < 4; ++gt)
                gv[gt] = sGate[b][chid + 32*gt] + b2r[gt];
            const float iv = fast_sigmoidf(gv[0]);
            const float fv = fast_sigmoidf(gv[1]);
            const float gg = fast_tanhf(gv[2]);
            const float ov = fast_sigmoidf(gv[3]);
            const float c  = fv*creg[p] + iv*gg;
            creg[p] = c;
            const float h  = ov * fast_tanhf(c);
            hh[p] = h;
            sH[b][chid] = (__bf16)h;
        }

        // output MLP via wave-level shuffle reduction:
        // lanes of this wave hold h2[b][0..31] for b = {cb0, cb1}
        float y0 = hh[0] * woutc;
        float y1 = hh[1] * woutc;
        #pragma unroll
        for (int m = 16; m >= 1; m >>= 1) {
            y0 += __shfl_xor(y0, m, 32);
            y1 += __shfl_xor(y1, m, 32);
        }
        if (lane == 0) {
            gy[(size_t)(bbase + cb0)*SEQ + t] = y0 + bout;
            gy[(size_t)(bbase + cb1)*SEQ + t] = y1 + bout;
        }
        __syncthreads();
    }
}

extern "C" void kernel_launch(void* const* d_in, const int* in_sizes, int n_in,
                              void* d_out, int out_size, void* d_ws, size_t ws_size,
                              hipStream_t stream) {
    (void)n_in; (void)out_size; (void)d_ws; (void)ws_size;
    const float* gx     = (const float*)d_in[0];
    const float* gsfc   = (const float*)d_in[1];
    const float* w_sfc1 = (const float*)d_in[2];
    const float* b_sfc1 = (const float*)d_in[3];
    const float* w_sfc2 = (const float*)d_in[4];
    const float* b_sfc2 = (const float*)d_in[5];
    const float* w_ih1  = (const float*)d_in[6];
    const float* w_hh1  = (const float*)d_in[7];
    const float* b_ih1  = (const float*)d_in[8];
    const float* b_hh1  = (const float*)d_in[9];
    const float* w_ih2  = (const float*)d_in[10];
    const float* w_hh2  = (const float*)d_in[11];
    const float* b_ih2  = (const float*)d_in[12];
    const float* b_hh2  = (const float*)d_in[13];
    const float* w_out  = (const float*)d_in[14];
    const float* b_out  = (const float*)d_in[15];
    float* gy = (float*)d_out;

    const int Btot = in_sizes[1] / NSFC_;   // 8192
    dim3 grid(Btot / TB), block(THREADS);
    lstm2_fused_kernel<<<grid, block, 0, stream>>>(
        gx, gsfc, w_sfc1, b_sfc1, w_sfc2, b_sfc2,
        w_ih1, w_hh1, b_ih1, b_hh1,
        w_ih2, w_hh2, b_ih2, b_hh2,
        w_out, b_out, gy);
}
